// DecoderRNN_25563645345796
// MI455X (gfx1250) — compile-verified, matrix operand order checked by reading, not run on hardware
//
#include <hip/hip_runtime.h>
#include <stdint.h>

#define HID  1024
#define BS   4096
#define TSTEPS 24
#define G3   3072            // 3*HID
#define KTOT 1056            // 1024 (h) + 32 (dec, 12 aux, 19 pad)
#define NEMB 5

typedef __attribute__((ext_vector_type(16))) __bf16 v16bf;
typedef __attribute__((ext_vector_type(8)))  float  v8f;

union FragU { uint4 u[2]; v16bf v; };

__device__ __forceinline__ unsigned short f2bf(float f) {
    unsigned int u = __float_as_uint(f);
    u += 0x7FFFu + ((u >> 16) & 1u);          // round-to-nearest-even
    return (unsigned short)(u >> 16);
}

// async copy global -> LDS (CDNA5), tracked by ASYNCcnt
__device__ __forceinline__ void async_ld_b128(uint32_t lds, const void* gptr) {
    asm volatile("global_load_async_to_lds_b128 %0, %1, off"
                 :: "v"(lds), "v"(gptr) : "memory");
}
__device__ __forceinline__ void wait_async0() {
    asm volatile("s_wait_asynccnt 0" ::: "memory");
}

// ---------------- one-time weight concat/convert: Wcat[j][k] bf16 ----------------
__global__ void build_wcat(const float* __restrict__ Whh, const float* __restrict__ Wih,
                           unsigned short* __restrict__ Wcat) {
    int idx = blockIdx.x * blockDim.x + threadIdx.x;
    if (idx >= G3 * KTOT) return;
    int j = idx / KTOT, k = idx % KTOT;
    float val = 0.f;
    if (k < HID)            val = Whh[j * HID + k];
    else if (k < HID + 13)  val = Wih[j * 13 + (k - HID)];
    Wcat[idx] = f2bf(val);
}

// ---------------- init: h0 into f32 hbuf and bf16 A-buffer, zero loss ----------------
__global__ void init_state(const float* __restrict__ hn, unsigned short* __restrict__ A0,
                           float* __restrict__ hbuf, float* __restrict__ lossAcc) {
    int idx = blockIdx.x * blockDim.x + threadIdx.x;
    if (idx == 0) *lossAcc = 0.f;
    if (idx >= BS * HID) return;
    int b = idx / HID, j = idx % HID;
    float h = hn[idx];
    hbuf[idx] = h;
    A0[(size_t)b * KTOT + j] = f2bf(h);
}

// ---------------- per-step: build x-chunk (dec | aux | pad) of A ----------------
__global__ void prep_x(const float* __restrict__ aux, const float* __restrict__ cp,
                       const float* __restrict__ tgt_seq, const float* __restrict__ v,
                       const float* __restrict__ prev_out, unsigned short* __restrict__ Acur,
                       int t) {
    int b = blockIdx.x * blockDim.x + threadIdx.x;
    if (b >= BS) return;
    float dec;
    if (t == 0) {
        dec = cp[b];
    } else {
        float tg = tgt_seq[b * TSTEPS + (t - 1)];          // tgt[:, t] = tgt_seq[:, t-1]
        float mk = (tg != 0.f) ? v[b * TSTEPS + t] : 0.f;  // mask
        dec = (mk == 1.f) ? tg : prev_out[b];
    }
    unsigned short* row = Acur + (size_t)b * KTOT + HID;
    row[0] = f2bf(dec);
#pragma unroll
    for (int f = 0; f < 12; ++f) row[1 + f] = f2bf(aux[(b * TSTEPS + t) * 12 + f]);
#pragma unroll
    for (int f = 13; f < 32; ++f) row[f] = 0;
}

// ---------------- fused GEMM (bf16 WMMA) + GRU cell ----------------
// block: 256 thr = 8 waves, tile 128 rows x 64 cols of h_new.
// wave grid 4x2 -> each wave 32 rows x 32 cols -> per K-chunk: 2 A-frags, 6 B-frags, 12 WMMA.
// Staging is pure async global->LDS (no VGPR round trip), double-buffered.
__global__ __launch_bounds__(256, 2) void gru_gemm(
    const unsigned short* __restrict__ Acur, unsigned short* __restrict__ Anxt,
    const unsigned short* __restrict__ Wcat, float* __restrict__ hbuf,
    const float* __restrict__ bih, const float* __restrict__ bhh) {

    __shared__ unsigned short At[2][128 * 32];   // 2 x 8 KB
    __shared__ unsigned short Bt[2][192 * 32];   // 2 x 12 KB  (3 gates x 64 cols)

    const int tid  = threadIdx.x;
    const int lane = tid & 31, wave = tid >> 5;
    const int wr = wave >> 1, wc = wave & 1;        // wave row/col in block
    const int rowBlock = blockIdx.y * 128;
    const int colBlock = blockIdx.x * 64;

    // ---- async staging geometry: 16B segments ----
    // A: 512 segs (128 rows x 4); thread handles seg tid and tid+256
    const int ar0 = tid >> 2;                 // 0..63
    const int ao  = (tid & 3) * 8;            // element offset within 32
    const unsigned short* aG0 = Acur + (size_t)(rowBlock + ar0) * KTOT + ao;
    const unsigned short* aG1 = Acur + (size_t)(rowBlock + ar0 + 64) * KTOT + ao;
    const uint32_t atBase = (uint32_t)(uintptr_t)(&At[0][0]);
    const uint32_t aL0 = atBase + (uint32_t)(ar0 * 32 + ao) * 2;
    const uint32_t aL1 = aL0 + 64 * 32 * 2;
    // B: 768 segs (192 rows x 4); thread handles segs tid, tid+256, tid+512
    const uint32_t btBase = (uint32_t)(uintptr_t)(&Bt[0][0]);
    const unsigned short* bG[3];
    uint32_t bL[3];
#pragma unroll
    for (int i = 0; i < 3; ++i) {
        int rb = (tid >> 2) + i * 64;         // 0..191
        int g = rb >> 6, c = rb & 63;
        bG[i] = Wcat + (size_t)(g * HID + colBlock + c) * KTOT + ao;
        bL[i] = btBase + (uint32_t)(rb * 32 + ao) * 2;
    }

    auto stage = [&](int kc, int buf) {
        const int ge = kc * 32;                       // global elem offset in K
        const uint32_t la = (uint32_t)buf * (128 * 32 * 2);
        const uint32_t lb = (uint32_t)buf * (192 * 32 * 2);
        async_ld_b128(aL0 + la, aG0 + ge);
        async_ld_b128(aL1 + la, aG1 + ge);
#pragma unroll
        for (int i = 0; i < 3; ++i)
            async_ld_b128(bL[i] + lb, bG[i] + ge);
    };

    // ---- accumulators: 3 gates combined + separate x-part for n-gate ----
    v8f acch[2][2][3];   // [rowtile][coltile][gate]  (r,z include x-chunk; n = h-part only)
    v8f accx[2][2];      // n-gate x-part (i_n)
    const v8f vzero = {0.f,0.f,0.f,0.f,0.f,0.f,0.f,0.f};
#pragma unroll
    for (int rt = 0; rt < 2; ++rt)
#pragma unroll
        for (int ct = 0; ct < 2; ++ct) {
#pragma unroll
            for (int g = 0; g < 3; ++g) acch[rt][ct][g] = vzero;
            accx[rt][ct] = vzero;
        }

    const int mn   = lane & 15;
    const int half = lane >> 4;

    auto loadA = [&](int buf, int rt, FragU& fa) {
        const unsigned short* ap = &At[buf][(wr * 32 + rt * 16 + mn) * 32];
        fa.u[0] = *(const uint4*)(ap + 8 * half);
        fa.u[1] = *(const uint4*)(ap + 16 + 8 * half);
    };
    auto loadB = [&](int buf, int g, int ct, FragU& fbv) {
        const unsigned short* bp = &Bt[buf][(g * 64 + wc * 32 + ct * 16 + mn) * 32];
        fbv.u[0] = *(const uint4*)(bp + 8 * half);
        fbv.u[1] = *(const uint4*)(bp + 16 + 8 * half);
    };

    stage(0, 0);
    wait_async0();
    __syncthreads();

    // main loop: chunks 0..31 (h-part), all 3 gates into acch
    for (int kc = 0; kc < 32; ++kc) {
        const int buf = kc & 1;
        stage(kc + 1, buf ^ 1);                        // includes x-chunk (kc==31 -> 32)

        FragU a0, a1;
        loadA(buf, 0, a0);
        loadA(buf, 1, a1);
#pragma unroll
        for (int ct = 0; ct < 2; ++ct) {
#pragma unroll
            for (int g = 0; g < 3; ++g) {
                FragU fbv;
                loadB(buf, g, ct, fbv);
                acch[0][ct][g] = __builtin_amdgcn_wmma_f32_16x16x32_bf16(
                    false, a0.v, false, fbv.v, (short)0, acch[0][ct][g], false, false);
                acch[1][ct][g] = __builtin_amdgcn_wmma_f32_16x16x32_bf16(
                    false, a1.v, false, fbv.v, (short)0, acch[1][ct][g], false, false);
            }
        }
        wait_async0();
        __syncthreads();
    }

    // peeled x-chunk (chunk 32, buffer 0): r,z -> acch ; n -> accx
    {
        const int buf = 0;
        FragU a0, a1;
        loadA(buf, 0, a0);
        loadA(buf, 1, a1);
#pragma unroll
        for (int ct = 0; ct < 2; ++ct) {
#pragma unroll
            for (int g = 0; g < 2; ++g) {              // r, z fold x-part in
                FragU fbv;
                loadB(buf, g, ct, fbv);
                acch[0][ct][g] = __builtin_amdgcn_wmma_f32_16x16x32_bf16(
                    false, a0.v, false, fbv.v, (short)0, acch[0][ct][g], false, false);
                acch[1][ct][g] = __builtin_amdgcn_wmma_f32_16x16x32_bf16(
                    false, a1.v, false, fbv.v, (short)0, acch[1][ct][g], false, false);
            }
            FragU fbn;                                 // n gate: keep i_n separate
            loadB(buf, 2, ct, fbn);
            accx[0][ct] = __builtin_amdgcn_wmma_f32_16x16x32_bf16(
                false, a0.v, false, fbn.v, (short)0, accx[0][ct], false, false);
            accx[1][ct] = __builtin_amdgcn_wmma_f32_16x16x32_bf16(
                false, a1.v, false, fbn.v, (short)0, accx[1][ct], false, false);
        }
    }

    // ---- GRU epilogue ----
#pragma unroll
    for (int ct = 0; ct < 2; ++ct) {
        const int j = colBlock + wc * 32 + ct * 16 + mn;
        const float bi0 = bih[j],           bh0 = bhh[j];
        const float bi1 = bih[HID + j],     bh1 = bhh[HID + j];
        const float bi2 = bih[2 * HID + j], bh2 = bhh[2 * HID + j];
#pragma unroll
        for (int rt = 0; rt < 2; ++rt) {
#pragma unroll
            for (int e = 0; e < 8; ++e) {
                int b = rowBlock + wr * 32 + rt * 16 + e + 8 * half;
                float sr = acch[rt][ct][0][e] + bi0 + bh0;
                float sz = acch[rt][ct][1][e] + bi1 + bh1;
                float r  = 1.f / (1.f + __expf(-sr));
                float z  = 1.f / (1.f + __expf(-sz));
                float nn = tanhf(accx[rt][ct][e] + bi2 + r * (acch[rt][ct][2][e] + bh2));
                float hp = hbuf[(size_t)b * HID + j];
                float hnew = (1.f - z) * nn + z * hp;
                hbuf[(size_t)b * HID + j] = hnew;
                Anxt[(size_t)b * KTOT + j] = f2bf(hnew);
            }
        }
    }
}

// ---------------- per-step head: logits vs emb, logsumexp loss, W_out projection ----------------
__global__ void head_kernel(const float* __restrict__ hbuf, const float* __restrict__ emb,
                            const float* __restrict__ wout, const float* __restrict__ bout,
                            const int* __restrict__ st, float* __restrict__ dout,
                            float* __restrict__ prev_out, float* __restrict__ lossAcc, int t) {
    __shared__ float red[8];
    const int lane = threadIdx.x & 31, wave = threadIdx.x >> 5;
    const int b = blockIdx.x * 8 + wave;
    float s[6] = {0.f, 0.f, 0.f, 0.f, 0.f, 0.f};
    for (int k = lane; k < HID; k += 32) {
        float h = hbuf[(size_t)b * HID + k];
#pragma unroll
        for (int c = 0; c < NEMB; ++c) s[c] += h * emb[c * HID + k];
        s[5] += h * wout[k];
    }
#pragma unroll
    for (int c = 0; c < 6; ++c)
        for (int off = 16; off > 0; off >>= 1) s[c] += __shfl_xor(s[c], off, 32);

    if (lane == 0) {
        float m = s[0];
#pragma unroll
        for (int c = 1; c < NEMB; ++c) m = fmaxf(m, s[c]);
        float se = 0.f;
#pragma unroll
        for (int c = 0; c < NEMB; ++c) se += __expf(s[c] - m);
        float lse = m + __logf(se);
        int pos = st[b * TSTEPS + t];
        float contrib = lse - s[pos];
        float out = s[5] + bout[0];
        dout[b * TSTEPS + t] = out;
        prev_out[b] = out;
        red[wave] = contrib;
    }
    __syncthreads();
    if (threadIdx.x == 0) {
        float c = 0.f;
#pragma unroll
        for (int w = 0; w < 8; ++w) c += red[w];
        atomicAdd(lossAcc, c * (1.0f / BS));
    }
}

__global__ void finalize(const float* __restrict__ lossAcc, float* __restrict__ dout) {
    if (threadIdx.x == 0 && blockIdx.x == 0) dout[BS * TSTEPS] = *lossAcc / (float)TSTEPS;
}

// ---------------- launcher ----------------
extern "C" void kernel_launch(void* const* d_in, const int* in_sizes, int n_in,
                              void* d_out, int out_size, void* d_ws, size_t ws_size,
                              hipStream_t stream) {
    const float* aux  = (const float*)d_in[0];
    const float* cp   = (const float*)d_in[1];
    const float* hn   = (const float*)d_in[2];
    const int*   st   = (const int*)d_in[3];
    const float* tgt  = (const float*)d_in[4];
    const float* v    = (const float*)d_in[5];
    const float* Wih  = (const float*)d_in[6];
    const float* Whh  = (const float*)d_in[7];
    const float* bih  = (const float*)d_in[8];
    const float* bhh  = (const float*)d_in[9];
    const float* emb  = (const float*)d_in[10];
    const float* wout = (const float*)d_in[11];
    const float* bout = (const float*)d_in[12];
    float* dout = (float*)d_out;

    char* ws = (char*)d_ws;
    size_t off = 0;
    unsigned short* Wcat = (unsigned short*)(ws + off); off += (size_t)G3 * KTOT * 2;
    unsigned short* A[2];
    A[0] = (unsigned short*)(ws + off); off += (size_t)BS * KTOT * 2;
    A[1] = (unsigned short*)(ws + off); off += (size_t)BS * KTOT * 2;
    float* hbuf = (float*)(ws + off);   off += (size_t)BS * HID * 4;
    float* prev = (float*)(ws + off);   off += (size_t)BS * 4;
    float* lossAcc = (float*)(ws + off); off += 256;

    build_wcat<<<(G3 * KTOT + 255) / 256, 256, 0, stream>>>(Whh, Wih, Wcat);
    init_state<<<(BS * HID + 255) / 256, 256, 0, stream>>>(hn, A[0], hbuf, lossAcc);

    for (int t = 0; t < TSTEPS; ++t) {
        unsigned short* Acur = A[t & 1];
        unsigned short* Anxt = A[(t + 1) & 1];
        prep_x<<<(BS + 255) / 256, 256, 0, stream>>>(aux, cp, tgt, v, prev, Acur, t);
        dim3 g(HID / 64, BS / 128);
        gru_gemm<<<g, 256, 0, stream>>>(Acur, Anxt, Wcat, hbuf, bih, bhh);
        head_kernel<<<BS / 8, 256, 0, stream>>>(hbuf, emb, wout, bout, st, dout, prev, lossAcc, t);
    }
    finalize<<<1, 1, 0, stream>>>(lossAcc, dout);
}
